// M3GNetMultiTask_74079595921436
// MI455X (gfx1250) — matrix operand order repaired and checked by reading, not compiled, not run on hardware
//
#include <hip/hip_runtime.h>
#include <hip/hip_bf16.h>
#include <math.h>

typedef __attribute__((ext_vector_type(16))) _Float16 v16h;
typedef __attribute__((ext_vector_type(8)))  float    v8f;

#define N_NODES 1000000
#define N_GRAPHS 8192
#define HIDDEN 128

// ---------------------------------------------------------------------------
// K0: zero pooled accumulator, init gmax/gsum, precompute v = w_proj @ w_att
//     and c0 = b_proj . w_att + b_att   (logit_i = x_i . v + c0)
// ---------------------------------------------------------------------------
__global__ void k_init(float* __restrict__ pooled, float* __restrict__ gmax,
                       float* __restrict__ gsum, float* __restrict__ vvec,
                       float* __restrict__ c0,
                       const float* __restrict__ w_proj,
                       const float* __restrict__ w_att,
                       const float* __restrict__ b_proj,
                       const float* __restrict__ b_att) {
  int tid = blockIdx.x * blockDim.x + threadIdx.x;
  int nthr = gridDim.x * blockDim.x;
  for (int i = tid; i < N_GRAPHS * HIDDEN; i += nthr) pooled[i] = 0.0f;
  if (blockIdx.x == 0) {
    if (threadIdx.x < 64) {
      float s = 0.0f;
      #pragma unroll 8
      for (int j = 0; j < HIDDEN; ++j) s += w_proj[threadIdx.x * HIDDEN + j] * w_att[j];
      vvec[threadIdx.x] = s;
    } else if (threadIdx.x == 64) {
      float s = 0.0f;
      for (int j = 0; j < HIDDEN; ++j) s += b_proj[j] * w_att[j];
      *c0 = s + b_att[0];
    } else if (threadIdx.x == 65) {
      *gmax = -3.402823e38f;
    } else if (threadIdx.x == 66) {
      *gsum = 0.0f;
    }
  }
}

// ---------------------------------------------------------------------------
// K0b: pre-swizzle weights into f16 WMMA B-fragment layout.
//  bpackP: w_proj  -> [nt(8)][frag(2)][lane(32)][e(16)]        (16 KB)
//  bpackH: {we1,ws1,wc1,wm1} -> [head(4)][nt(8)][f(8)][lane(32)][e(16)] (256 KB)
//  Fragment element e on lane L holds B[k = kbase + (L>=16)*8 + (e&7) + (e>>3)*16][n = nt*16 + (L&15)].
// ---------------------------------------------------------------------------
__global__ void k_pack(const float* __restrict__ w_proj,
                       const float* __restrict__ we1, const float* __restrict__ ws1,
                       const float* __restrict__ wc1, const float* __restrict__ wm1,
                       _Float16* __restrict__ bpackP, _Float16* __restrict__ bpackH) {
  int i = blockIdx.x * 256 + threadIdx.x;
  if (i < 8192) {
    int e = i & 15, lane = (i >> 4) & 31, f = (i >> 9) & 1, nt = i >> 10;
    int k = f * 32 + (lane >> 4) * 8 + (e & 7) + (e >> 3) * 16;
    int n = nt * 16 + (lane & 15);
    bpackP[i] = (_Float16)w_proj[k * HIDDEN + n];
  }
  if (i < 131072) {
    int e = i & 15, lane = (i >> 4) & 31, f = (i >> 9) & 7, nt = (i >> 12) & 7, head = i >> 15;
    int k = f * 32 + (lane >> 4) * 8 + (e & 7) + (e >> 3) * 16;
    int n = nt * 16 + (lane & 15);
    const float* w1 = (head == 0) ? we1 : (head == 1) ? ws1 : (head == 2) ? wc1 : wm1;
    bpackH[i] = (_Float16)w1[(size_t)k * HIDDEN + n];
  }
}

// ---------------------------------------------------------------------------
// K1: logits[i] = x_i . v + c0 ; per-block max
// ---------------------------------------------------------------------------
__global__ void k_logits(const float* __restrict__ x, const float* __restrict__ vvec,
                         const float* __restrict__ c0, float* __restrict__ logits,
                         float* __restrict__ blockmax, int n) {
  __shared__ float red[256];
  int i = blockIdx.x * 256 + threadIdx.x;
  float lg = -3.402823e38f;
  if (i < n) {
    const float* xr = x + (size_t)i * 64;
    float s = 0.0f;
    #pragma unroll
    for (int k = 0; k < 64; k += 4) {
      float4 a = *(const float4*)(xr + k);
      float4 b = *(const float4*)(vvec + k);
      s += a.x * b.x + a.y * b.y + a.z * b.z + a.w * b.w;
    }
    s += *c0;
    logits[i] = s;
    lg = s;
  }
  red[threadIdx.x] = lg;
  __syncthreads();
  #pragma unroll
  for (int off = 128; off > 0; off >>= 1) {
    if (threadIdx.x < off) red[threadIdx.x] = fmaxf(red[threadIdx.x], red[threadIdx.x + off]);
    __syncthreads();
  }
  if (threadIdx.x == 0) blockmax[blockIdx.x] = red[0];
}

// ---------------------------------------------------------------------------
// K2: reduce per-block maxima -> gmax (single block)
// ---------------------------------------------------------------------------
__global__ void k_maxreduce(const float* __restrict__ blockmax, int nb,
                            float* __restrict__ gmax) {
  __shared__ float red[256];
  float m = -3.402823e38f;
  for (int i = threadIdx.x; i < nb; i += 256) m = fmaxf(m, blockmax[i]);
  red[threadIdx.x] = m;
  __syncthreads();
  #pragma unroll
  for (int off = 128; off > 0; off >>= 1) {
    if (threadIdx.x < off) red[threadIdx.x] = fmaxf(red[threadIdx.x], red[threadIdx.x + off]);
    __syncthreads();
  }
  if (threadIdx.x == 0) *gmax = red[0];
}

// ---------------------------------------------------------------------------
// K3: gsum = sum_i exp(logit_i - gmax)
// ---------------------------------------------------------------------------
__global__ void k_sumexp(const float* __restrict__ logits, const float* __restrict__ gmax,
                         float* __restrict__ gsum, int n) {
  __shared__ float red[256];
  float m = *gmax;
  float s = 0.0f;
  for (int i = blockIdx.x * 256 + threadIdx.x; i < n; i += gridDim.x * 256)
    s += __expf(logits[i] - m);
  red[threadIdx.x] = s;
  __syncthreads();
  #pragma unroll
  for (int off = 128; off > 0; off >>= 1) {
    if (threadIdx.x < off) red[threadIdx.x] += red[threadIdx.x + off];
    __syncthreads();
  }
  if (threadIdx.x == 0) atomicAdd(gsum, red[0]);
}

// ---------------------------------------------------------------------------
// K4: main node pass. Per wave: one 16-node tile. h = x@Wp + b via WMMA f16
//     (B fragments pre-packed f16: 2x b128 loads each, no converts),
//     weight by alpha_i, segment-sum into pooled[batch]. batch is sorted, so
//     fast path = whole tile in one graph (shuffle-combine, 1 atomic/column).
//     Boundary-tile graph offsets hoisted as 32-bit ints (low VGPR cost,
//     keeps global address space for the atomics); nt loop kept rolled so
//     the A fragments stay resident in f16 registers.
// ---------------------------------------------------------------------------
__global__ void __launch_bounds__(256)
k_nodes(const float* __restrict__ x, const int* __restrict__ batch,
        const float* __restrict__ logits, const _Float16* __restrict__ bpackP,
        const float* __restrict__ b_proj, const float* __restrict__ gmax,
        const float* __restrict__ gsum, float* __restrict__ pooled, int ntiles) {
  int wave = threadIdx.x >> 5;
  int lane = threadIdx.x & 31;
  int tile = blockIdx.x * 8 + wave;
  if (tile >= ntiles) return;       // wave-uniform exit; EXEC full for WMMA

  int row   = lane & 15;
  int khalf = lane >> 4;            // 0 or 1
  int node0 = tile * 16;

  float invZ = 1.0f / (*gsum);
  float gm = *gmax;
  float wl = 0.0f;
  if (lane < 16) wl = __expf(logits[node0 + lane] - gm) * invZ;
  int bl = batch[node0 + row];
  int b0  = __shfl(bl, 0, 32);
  int b15 = __shfl(bl, 15, 32);
  bool uniform = (b0 == b15);       // sorted batch => all 16 rows same graph
  float wrow[8];
  #pragma unroll
  for (int r = 0; r < 8; ++r) wrow[r] = __shfl(wl, khalf * 8 + r, 32);

  // Fallback-only: per-row pooled row offsets (loop-invariant over nt)
  int goff[8];
  #pragma unroll
  for (int r = 0; r < 8; ++r) goff[r] = 0;
  if (!uniform) {
    #pragma unroll
    for (int r = 0; r < 8; ++r)
      goff[r] = __shfl(bl, khalf * 8 + r, 32) * HIDDEN;
  }

  // A fragments (16x32 f16): lane row=row, K runs {klo..klo+7, klo+16..klo+23}
  const float* xa = x + (size_t)(node0 + row) * 64 + khalf * 8;
  v16h a0, a1;
  #pragma unroll
  for (int e = 0; e < 8; ++e) {
    a0[e]     = (_Float16)xa[e];
    a0[e + 8] = (_Float16)xa[16 + e];
    a1[e]     = (_Float16)xa[32 + e];
    a1[e + 8] = (_Float16)xa[48 + e];
  }

  const v16h* bp = (const v16h*)bpackP;   // [nt][frag][lane] pre-swizzled
  #pragma unroll 1
  for (int nt = 0; nt < 8; ++nt) {
    v16h bf0 = bp[(nt * 2 + 0) * 32 + lane];
    v16h bf1 = bp[(nt * 2 + 1) * 32 + lane];
    int ncol = nt * 16 + row;
    float bias = b_proj[ncol];
    v8f c8;
    #pragma unroll
    for (int r = 0; r < 8; ++r) c8[r] = bias;   // bias is weighted per-node too
    c8 = __builtin_amdgcn_wmma_f32_16x16x32_f16(false, a0, false, bf0, (short)0, c8, false, false);
    c8 = __builtin_amdgcn_wmma_f32_16x16x32_f16(false, a1, false, bf1, (short)0, c8, false, false);

    if (uniform) {
      float s = 0.0f;
      #pragma unroll
      for (int r = 0; r < 8; ++r) s += wrow[r] * c8[r];
      s += __shfl_xor(s, 16, 32);   // lane L and L^16 hold same column
      if (lane < 16) atomicAdd(&pooled[(size_t)b0 * HIDDEN + nt * 16 + lane], s);
    } else {
      #pragma unroll
      for (int r = 0; r < 8; ++r)
        atomicAdd(&pooled[goff[r] + ncol], wrow[r] * c8[r]);
    }
  }
}

// ---------------------------------------------------------------------------
// K5: fused head layer-1: act[8192,512] = relu([pooled|sg] @ [we1|ws1|wc1|wm1] + b)
//     One block per 16-graph tile; each wave does 4 of the 32 column tiles.
//     B fragments come pre-packed f16 from bpackH.
// ---------------------------------------------------------------------------
__global__ void __launch_bounds__(256)
k_heads1(const float* __restrict__ pooled, const float* __restrict__ sg_table,
         const int* __restrict__ space_group, const _Float16* __restrict__ bpackH,
         const float* __restrict__ be1, const float* __restrict__ bs1,
         const float* __restrict__ bc1, const float* __restrict__ bm1,
         float* __restrict__ act) {
  int wave = threadIdx.x >> 5;
  int lane = threadIdx.x & 31;
  int row   = lane & 15;
  int khalf = lane >> 4;
  int mtile = blockIdx.x;
  int g = mtile * 16 + row;
  int sgrow = space_group[g];

  // A fragments over K=256: f=0..3 from pooled, f=4..7 from sg_table gather
  v16h afr[8];
  #pragma unroll
  for (int f = 0; f < 8; ++f) {
    const float* src = (f < 4) ? (pooled + (size_t)g * HIDDEN)
                               : (sg_table + (size_t)sgrow * HIDDEN);
    int klocal = (f & 3) * 32 + khalf * 8;
    #pragma unroll
    for (int e = 0; e < 8; ++e) {
      afr[f][e]     = (_Float16)src[klocal + e];
      afr[f][e + 8] = (_Float16)src[klocal + 16 + e];
    }
  }

  const float* b1s[4] = {be1, bs1, bc1, bm1};
  const v16h* bh = (const v16h*)bpackH;   // [head][nt][f][lane] pre-swizzled

  #pragma unroll
  for (int q = 0; q < 4; ++q) {
    int nt = wave * 4 + q;           // 0..31 over 512 fused columns
    int head = nt >> 3;              // 128 cols per head
    int ncol_local = (nt & 7) * 16 + row;
    float bias = b1s[head][ncol_local];
    v8f c8;
    #pragma unroll
    for (int r = 0; r < 8; ++r) c8[r] = bias;
    #pragma unroll
    for (int f = 0; f < 8; ++f) {
      v16h bfr = bh[((size_t)nt * 8 + f) * 32 + lane];  // nt already encodes head*8+(nt&7)
      c8 = __builtin_amdgcn_wmma_f32_16x16x32_f16(false, afr[f], false, bfr, (short)0, c8, false, false);
    }
    #pragma unroll
    for (int r = 0; r < 8; ++r) {
      int m = mtile * 16 + khalf * 8 + r;
      act[(size_t)m * 512 + nt * 16 + row] = fmaxf(c8[r], 0.0f);
    }
  }
}

// ---------------------------------------------------------------------------
// K6: tiny second layers (64/3/7/3 outputs), one thread per output element
// ---------------------------------------------------------------------------
__global__ void k_heads2(const float* __restrict__ act,
                         const float* __restrict__ we2, const float* __restrict__ be2,
                         const float* __restrict__ ws2, const float* __restrict__ bs2,
                         const float* __restrict__ wc2, const float* __restrict__ bc2,
                         const float* __restrict__ wm2, const float* __restrict__ bm2,
                         float* __restrict__ out) {
  int idx = blockIdx.x * 256 + threadIdx.x;
  if (idx >= N_GRAPHS * 77) return;
  int g = idx / 77, j = idx % 77;
  const float *w2, *b2;
  int coloff, nw, jj;
  size_t outoff;
  if (j < 64)      { w2 = we2; b2 = be2; coloff = 0;   nw = 64; jj = j;      outoff = 0; }
  else if (j < 67) { w2 = ws2; b2 = bs2; coloff = 128; nw = 3;  jj = j - 64; outoff = (size_t)N_GRAPHS * 64; }
  else if (j < 74) { w2 = wc2; b2 = bc2; coloff = 256; nw = 7;  jj = j - 67; outoff = (size_t)N_GRAPHS * 67; }
  else             { w2 = wm2; b2 = bm2; coloff = 384; nw = 3;  jj = j - 74; outoff = (size_t)N_GRAPHS * 74; }
  const float* ar = act + (size_t)g * 512 + coloff;
  float s = b2[jj];
  #pragma unroll 4
  for (int k = 0; k < HIDDEN; ++k) s += ar[k] * w2[(size_t)k * nw + jj];
  out[outoff + (size_t)g * nw + jj] = s;
}

// ---------------------------------------------------------------------------
extern "C" void kernel_launch(void* const* d_in, const int* in_sizes, int n_in,
                              void* d_out, int out_size, void* d_ws, size_t ws_size,
                              hipStream_t stream) {
  const float* x           = (const float*)d_in[0];
  const int*   batch       = (const int*)  d_in[1];
  const int*   space_group = (const int*)  d_in[2];
  const float* w_proj      = (const float*)d_in[3];
  const float* b_proj      = (const float*)d_in[4];
  const float* w_att       = (const float*)d_in[5];
  const float* b_att       = (const float*)d_in[6];
  const float* sg_table    = (const float*)d_in[7];
  const float* we1 = (const float*)d_in[8];  const float* be1 = (const float*)d_in[9];
  const float* we2 = (const float*)d_in[10]; const float* be2 = (const float*)d_in[11];
  const float* ws1 = (const float*)d_in[12]; const float* bs1 = (const float*)d_in[13];
  const float* ws2 = (const float*)d_in[14]; const float* bs2 = (const float*)d_in[15];
  const float* wc1 = (const float*)d_in[16]; const float* bc1 = (const float*)d_in[17];
  const float* wc2 = (const float*)d_in[18]; const float* bc2 = (const float*)d_in[19];
  const float* wm1 = (const float*)d_in[20]; const float* bm1 = (const float*)d_in[21];
  const float* wm2 = (const float*)d_in[22]; const float* bm2 = (const float*)d_in[23];
  float* out = (float*)d_out;
  float* ws  = (float*)d_ws;

  // workspace layout (floats): ~25.3 MB total
  float* logits   = ws;                         // 1,000,000
  float* blockmax = ws + 1000000;               // 4,096 (padded)
  float* gmax     = ws + 1004096;               // 1
  float* gsum     = ws + 1004097;               // 1
  float* vvec     = ws + 1004104;               // 64 (16B aligned)
  float* c0       = ws + 1004168;               // 1
  float* pooled   = ws + 1004176;               // 8192*128 = 1,048,576
  float* act      = pooled + N_GRAPHS * HIDDEN; // 8192*512 = 4,194,304
  _Float16* bpackP = (_Float16*)(act + (size_t)N_GRAPHS * 512); // 8,192 halves (32B aligned)
  _Float16* bpackH = bpackP + 8192;                             // 131,072 halves

  k_init<<<2048, 256, 0, stream>>>(pooled, gmax, gsum, vvec, c0, w_proj, w_att, b_proj, b_att);
  k_pack<<<512, 256, 0, stream>>>(w_proj, we1, ws1, wc1, wm1, bpackP, bpackH);

  int nb1 = (N_NODES + 255) / 256;
  k_logits<<<nb1, 256, 0, stream>>>(x, vvec, c0, logits, blockmax, N_NODES);
  k_maxreduce<<<1, 256, 0, stream>>>(blockmax, nb1, gmax);
  k_sumexp<<<1024, 256, 0, stream>>>(logits, gmax, gsum, N_NODES);

  int ntiles = N_NODES / 16;                  // 62,500 (N divisible by 16)
  k_nodes<<<(ntiles + 7) / 8, 256, 0, stream>>>(x, batch, logits, bpackP, b_proj,
                                                gmax, gsum, pooled, ntiles);

  k_heads1<<<N_GRAPHS / 16, 256, 0, stream>>>(pooled, sg_table, space_group, bpackH,
                                              be1, bs1, bc1, bm1, act);
  k_heads2<<<(N_GRAPHS * 77 + 255) / 256, 256, 0, stream>>>(act, we2, be2, ws2, bs2,
                                                            wc2, bc2, wm2, bm2, out);
}
